// HyperConvEmbedder_55997783605368
// MI455X (gfx1250) — matrix operand order repaired and checked by reading, not compiled, not run on hardware
//
#include <hip/hip_runtime.h>
#include <hip/hip_fp16.h>
#include <cstdint>

// Problem constants (from reference)
#define NQ    100000      // queried nodes
#define NE    800000      // membership edges
#define MEMD  128
#define CAT   256         // MEM_D + TIME_D
#define EMBD  128
#define TILES (NQ / 16)   // 6250 row tiles of 16
#define LDA   264         // padded f16 row stride (528B row: 16B aligned, bank-clean)

typedef __attribute__((ext_vector_type(16))) _Float16 v16h;
typedef __attribute__((ext_vector_type(8)))  _Float16 v8h;
typedef __attribute__((ext_vector_type(4)))  _Float16 v4h;
typedef __attribute__((ext_vector_type(8)))  float    v8f;

// ---------------------------------------------------------------------------
__device__ __forceinline__ void wave_sync() {
  __builtin_amdgcn_fence(__ATOMIC_ACQ_REL, "wavefront");
  __builtin_amdgcn_wave_barrier();
}

// A fragment: 16x32 f16 tile (ISA 7.12.2 layout); 2x ds_load_b128 per lane.
__device__ __forceinline__ v16h load_a_frag(const _Float16* As, int lda) {
  const int lane = threadIdx.x & 31;
  const int m = lane & 15, half = lane >> 4;
  const _Float16* p = As + m * lda + 8 * half;
  v16h a;
  v8h lo = *(const v8h*)(p);        // K = 8*half .. +7
  v8h hi = *(const v8h*)(p + 16);   // K = 16 + 8*half .. +7
#pragma unroll
  for (int e = 0; e < 8; ++e) { a[e] = lo[e]; a[8 + e] = hi[e]; }
  return a;
}

// B fragment from pre-swizzled LDS weights: one contiguous 32B chunk per lane.
__device__ __forceinline__ v16h load_b_fragS(const _Float16* WS, int ks, int nt) {
  const int lane = threadIdx.x & 31;
  return *(const v16h*)(WS + ((((ks << 3) + nt) << 5) + lane) * 16);
}

// One [16,256]x[256,16] column-tile with software-pipelined B loads.
// Only a single v8f accumulator live -> no spills.
__device__ __forceinline__ v8f gemm_tile16(const v16h af[8], const _Float16* WS,
                                           int nt) {
  v8f c = {};
  v16h bcur = load_b_fragS(WS, 0, nt);
#pragma unroll
  for (int ks = 0; ks < 8; ++ks) {
    v16h bnext = (ks < 7) ? load_b_fragS(WS, ks + 1, nt) : bcur;
    c = __builtin_amdgcn_wmma_f32_16x16x32_f16(
        false, af[ks], false, bcur, (short)0, c, false, false);
    bcur = bnext;
  }
  return c;
}

// Stage one [256,128] f32 weight into LDS as f16, swizzled to B-fragment order:
// dst[((ks*8+nt)*32 + lane)*16 + e] ; k = ks*32 + 8*half + e + (e >= 8 ? 8 : 0).
__device__ __forceinline__ void stage_weight(_Float16* dst, const float* src,
                                             int tid, int nthreads) {
  for (int o = tid; o < 32768; o += nthreads) {
    const int e = o & 15, ln = (o >> 4) & 31, f = o >> 9;
    const int ks = f >> 3, nt = f & 7;
    const int n = ln & 15, half = ln >> 4;
    const int v = e >> 1, odd = e & 1;
    const int k = ks * 32 + 8 * half + (2 * v + odd) + (((2 * v + odd) >> 3) << 3);
    const int col = nt * 16 + n;
    dst[o] = (_Float16)src[k * EMBD + col];
  }
}

// ---------------------------------------------------------------------------
// Kernel 0: zero workspace (acc[NQ,256] f16 + sumw[NQ] f32, contiguous).
__global__ void k_zero(float4* __restrict__ p, long n4) {
  long i = (long)blockIdx.x * blockDim.x + threadIdx.x;
  const long stride = (long)gridDim.x * blockDim.x;
  const float4 z = {0.f, 0.f, 0.f, 0.f};
  for (; i < n4; i += stride) p[i] = z;
}

// ---------------------------------------------------------------------------
// Kernel 1: per-edge scatter of weighted concat(c_mem, time_enc) into f16 acc
// via packed-f16 atomics (global_atomic_pk_add_f16). acc (51 MB) is
// L2-resident on MI455X (192 MB L2) so the RMW traffic never touches HBM.
__global__ void k_edge_scatter(const float* __restrict__ c_mem,
                               const float* __restrict__ c_last,
                               const float* __restrict__ t,
                               const float* __restrict__ mem_vals,
                               const float* __restrict__ time_w,
                               const float* __restrict__ time_b,
                               const int*   __restrict__ n_idx,
                               const int*   __restrict__ c_idx,
                               __half* __restrict__ acc,
                               float*  __restrict__ sumw) {
  const int lane = threadIdx.x & 31;
  const int wv = (int)((blockIdx.x * blockDim.x + threadIdx.x) >> 5);
  const int nw = (int)((gridDim.x * blockDim.x) >> 5);
  const float4 tw = ((const float4*)time_w)[lane];
  const float4 tb = ((const float4*)time_b)[lane];

  for (int e = wv; e < NE; e += nw) {
    const int   nn = n_idx[e];
    const int   cc = c_idx[e];
    const float w  = mem_vals[e];

    const int e2 = e + nw;                 // prefetch next community row
    if (e2 < NE) {
      const int c2 = c_idx[e2];
      __builtin_prefetch(c_mem + (size_t)c2 * MEMD + lane * 4, 0, 1);
    }

    const float  dt = t[nn] - c_last[cc];
    const float4 m4 = ((const float4*)(c_mem + (size_t)cc * MEMD))[lane];

    __half2* am = (__half2*)(acc + (size_t)nn * CAT + lane * 4);
    atomicAdd(am + 0, __floats2half2_rn(w * m4.x, w * m4.y));
    atomicAdd(am + 1, __floats2half2_rn(w * m4.z, w * m4.w));

    __half2* at = am + (MEMD / 2);
    atomicAdd(at + 0, __floats2half2_rn(w * __cosf(fmaf(dt, tw.x, tb.x)),
                                        w * __cosf(fmaf(dt, tw.y, tb.y))));
    atomicAdd(at + 1, __floats2half2_rn(w * __cosf(fmaf(dt, tw.z, tb.z)),
                                        w * __cosf(fmaf(dt, tw.w, tb.w))));

    if (lane == 0) atomicAdd(sumw + nn, w);
  }
}

// ---------------------------------------------------------------------------
// Kernel 2a: node + community WMMA phases; writes f16 concat CH[NQ,256].
// Stages only Wn+Wc (128 KB) -> 8 waves/block (2 waves/SIMD32).
#define SMEM2A_STAGE_OFF 131072               // 2 * 64KB swizzled weights
#define STAGE2A_BYTES    8512                 // A(8448) + sumw16(64)
#define WAVES2A          8
#define SMEM2A_TOTAL     (SMEM2A_STAGE_OFF + WAVES2A * STAGE2A_BYTES)  // 199168

__global__ void k_embed_branches(const float* __restrict__ n_mem,
                                 const float* __restrict__ n_last,
                                 const float* __restrict__ t,
                                 const float* __restrict__ time_w,
                                 const float* __restrict__ time_b,
                                 const float* __restrict__ Wn,
                                 const float* __restrict__ bn,
                                 const float* __restrict__ Wc,
                                 const float* __restrict__ bc,
                                 const int*   __restrict__ idx,
                                 const __half* __restrict__ acc,
                                 const float* __restrict__ sumw,
                                 __half* __restrict__ CH) {
  extern __shared__ char smem[];
  _Float16* WnS = (_Float16*)smem;
  _Float16* WcS = WnS + 32768;
  char*     stage = smem + SMEM2A_STAGE_OFF;

  stage_weight(WnS, Wn, threadIdx.x, blockDim.x);
  stage_weight(WcS, Wc, threadIdx.x, blockDim.x);
  __syncthreads();

  const int lane = threadIdx.x & 31;
  const int wave = threadIdx.x >> 5;
  _Float16* A    = (_Float16*)(stage + wave * STAGE2A_BYTES);  // 16 x LDA
  float*    sw16 = (float*)(stage + wave * STAGE2A_BYTES + 8448);

  const float4 tw = ((const float4*)time_w)[lane];
  const float4 tb = ((const float4*)time_b)[lane];
  const int ncol = lane & 15, half = lane >> 4;

  // Tile-invariant bias registers (one per N-tile, this lane's column).
  float bnr[8], bcr[8];
#pragma unroll
  for (int nt = 0; nt < 8; ++nt) {
    bnr[nt] = bn[nt * 16 + ncol];
    bcr[nt] = bc[nt * 16 + ncol];
  }

  const int gwave  = wave + blockIdx.x * WAVES2A;
  const int nwaves = gridDim.x * WAVES2A;

  for (int tile = gwave; tile < TILES; tile += nwaves) {
    const int base = tile << 4;

    // ---- Build node-branch A: [n_mem[idx] | cos(dt*w+b)] in f16 ----
#pragma unroll 1
    for (int r = 0; r < 16; ++r) {
      const int i  = base + r;
      const int id = idx[i];
      const float4 m4 = ((const float4*)(n_mem + (size_t)id * MEMD))[lane];
      _Float16* row = A + r * LDA;
      v4h pm = {(_Float16)m4.x, (_Float16)m4.y, (_Float16)m4.z, (_Float16)m4.w};
      *(v4h*)(row + lane * 4) = pm;
      const float dt = t[i] - n_last[id];
      v4h pt = {(_Float16)(__cosf(fmaf(dt, tw.x, tb.x))),
                (_Float16)(__cosf(fmaf(dt, tw.y, tb.y))),
                (_Float16)(__cosf(fmaf(dt, tw.z, tb.z))),
                (_Float16)(__cosf(fmaf(dt, tw.w, tb.w)))};
      *(v4h*)(row + MEMD + lane * 4) = pt;
    }
    wave_sync();

    // ---- nh = A @ Wn + bn -> CH[:, 0:128] ----
    {
      v16h af[8];
#pragma unroll
      for (int ks = 0; ks < 8; ++ks) af[ks] = load_a_frag(A + ks * 32, LDA);
#pragma unroll
      for (int nt = 0; nt < 8; ++nt) {
        v8f c = gemm_tile16(af, WnS, nt);
#pragma unroll
        for (int v = 0; v < 8; ++v) {
          const int m = v + 8 * half;
          CH[(size_t)(base + m) * CAT + nt * 16 + ncol] =
              (__half)(_Float16)(c[v] + bnr[nt]);
        }
      }
    }
    wave_sync();

    // ---- Build community-branch A from L2-resident f16 acc ----
#pragma unroll 1
    for (int r = 0; r < 16; ++r) {
      const int i = base + r;
      const uint4 chunk = ((const uint4*)(acc + (size_t)i * CAT))[lane];
      *(uint4*)(A + r * LDA + lane * 8) = chunk;
    }
    if (lane < 16) sw16[lane] = sumw[base + lane];
    wave_sync();

    // ---- cagg = A @ Wc + sumw*bc -> CH[:, 128:256] ----
    {
      v16h af[8];
#pragma unroll
      for (int ks = 0; ks < 8; ++ks) af[ks] = load_a_frag(A + ks * 32, LDA);
#pragma unroll
      for (int nt = 0; nt < 8; ++nt) {
        v8f c = gemm_tile16(af, WcS, nt);
#pragma unroll
        for (int v = 0; v < 8; ++v) {
          const int m = v + 8 * half;
          CH[(size_t)(base + m) * CAT + 128 + nt * 16 + ncol] =
              (__half)(_Float16)(c[v] + sw16[m] * bcr[nt]);
        }
      }
    }
    wave_sync();
  }
}

// ---------------------------------------------------------------------------
// Kernel 2b: out = CH @ Wo + bo (f32). Stages only Wo (64 KB) -> 8 waves/block.
#define SMEM2B_STAGE_OFF 65536
#define STAGE2B_BYTES    8448                 // A only
#define WAVES2B          8
#define SMEM2B_TOTAL     (SMEM2B_STAGE_OFF + WAVES2B * STAGE2B_BYTES)  // 133120

__global__ void k_embed_out(const float* __restrict__ Wo,
                            const float* __restrict__ bo,
                            const __half* __restrict__ CH,
                            float* __restrict__ out) {
  extern __shared__ char smem[];
  _Float16* WoS = (_Float16*)smem;
  char*     stage = smem + SMEM2B_STAGE_OFF;

  stage_weight(WoS, Wo, threadIdx.x, blockDim.x);
  __syncthreads();

  const int lane = threadIdx.x & 31;
  const int wave = threadIdx.x >> 5;
  _Float16* A = (_Float16*)(stage + wave * STAGE2B_BYTES);  // 16 x LDA

  const int ncol = lane & 15, half = lane >> 4;
  float bor[8];
#pragma unroll
  for (int nt = 0; nt < 8; ++nt) bor[nt] = bo[nt * 16 + ncol];

  const int gwave  = wave + blockIdx.x * WAVES2B;
  const int nwaves = gridDim.x * WAVES2B;

  for (int tile = gwave; tile < TILES; tile += nwaves) {
    const int base = tile << 4;

#pragma unroll 1
    for (int r = 0; r < 16; ++r) {
      const int i = base + r;
      const uint4 chunk = ((const uint4*)(CH + (size_t)i * CAT))[lane];
      *(uint4*)(A + r * LDA + lane * 8) = chunk;
    }
    wave_sync();

    v16h af[8];
#pragma unroll
    for (int ks = 0; ks < 8; ++ks) af[ks] = load_a_frag(A + ks * 32, LDA);
#pragma unroll
    for (int nt = 0; nt < 8; ++nt) {
      v8f c = gemm_tile16(af, WoS, nt);
#pragma unroll
      for (int v = 0; v < 8; ++v) {
        const int m = v + 8 * half;
        out[(size_t)(base + m) * EMBD + nt * 16 + ncol] = c[v] + bor[nt];
      }
    }
    wave_sync();
  }
}

// ---------------------------------------------------------------------------
extern "C" void kernel_launch(void* const* d_in, const int* in_sizes, int n_in,
                              void* d_out, int out_size, void* d_ws, size_t ws_size,
                              hipStream_t stream) {
  (void)in_sizes; (void)n_in; (void)out_size; (void)ws_size;
  const float* n_mem    = (const float*)d_in[0];
  const float* n_last   = (const float*)d_in[1];
  const float* c_mem    = (const float*)d_in[2];
  const float* c_last   = (const float*)d_in[3];
  const float* t        = (const float*)d_in[4];
  const float* mem_vals = (const float*)d_in[5];
  const float* time_w   = (const float*)d_in[6];
  const float* time_b   = (const float*)d_in[7];
  const float* Wn       = (const float*)d_in[8];
  const float* bn       = (const float*)d_in[9];
  const float* Wc       = (const float*)d_in[10];
  const float* bc       = (const float*)d_in[11];
  const float* Wo       = (const float*)d_in[12];
  const float* bo       = (const float*)d_in[13];
  const int*   idx      = (const int*)d_in[14];
  const int*   n_idx    = (const int*)d_in[15];
  const int*   c_idx    = (const int*)d_in[16];

  float*  out  = (float*)d_out;
  __half* acc  = (__half*)d_ws;                                  // [NQ,256] f16
  float*  sumw = (float*)((char*)d_ws + (size_t)NQ * CAT * 2);   // [NQ] f32
  __half* CH   = (__half*)((char*)d_ws + (size_t)NQ * (CAT * 2 + 4)); // [NQ,256] f16

  // Zero acc+sumw (contiguous NQ*516 bytes). CH is fully overwritten by k2a.
  const long n4 = (long)NQ * 516 / 16;
  k_zero<<<4096, 256, 0, stream>>>((float4*)d_ws, n4);

  k_edge_scatter<<<2048, 256, 0, stream>>>(c_mem, c_last, t, mem_vals,
                                           time_w, time_b, n_idx, c_idx,
                                           acc, sumw);

  k_embed_branches<<<256, 32 * WAVES2A, SMEM2A_TOTAL, stream>>>(
      n_mem, n_last, t, time_w, time_b, Wn, bn, Wc, bc,
      idx, acc, sumw, CH);

  k_embed_out<<<256, 32 * WAVES2B, SMEM2B_TOTAL, stream>>>(Wo, bo, CH, out);
}
